// MessagePassingReactionModel_9818295239491
// MI455X (gfx1250) — compile-verified, matrix operand order checked by reading, not run on hardware
//
#include <hip/hip_runtime.h>
#include <math.h>
#include <stdint.h>

// ---------------------------------------------------------------------------
// Equivariant GNN (e3nn-style) for MI455X / gfx1250.
// - Radial MLP runs on v_wmma_f32_16x16x32_f16, fused with the tensor-product
//   message kernel (per-edge weights live only in LDS).
// - A/B operands pre-permuted into WMMA lane layout -> contiguous b128 loads.
// - h[src] gather staged via global_load_async_to_lds_b128 (ASYNCcnt path).
// - Wigner-3j tensors computed on-device in fp64 each launch (deterministic).
// ---------------------------------------------------------------------------

typedef _Float16 half_t;
typedef __attribute__((ext_vector_type(16))) _Float16 v16h;
typedef __attribute__((ext_vector_type(8)))  float    v8f;

#define NNODES      5000
#define NGRAPHS     50
#define FEATDIM     36
#define NBASIS      10
#define NPATHS      11
#define MAXR        2.0f
#define W3J_TOTAL   363

static inline int cdiv(int a, int b) { return (a + b - 1) / b; }

// path tables (l1,l2,l3) in reference generation order
__device__ const int PL1[NPATHS]  = {0,0,0,1,1,1,1,2,2,2,2};
__device__ const int PL2[NPATHS]  = {0,1,2,0,1,1,2,0,1,2,2};
__device__ const int PL3[NPATHS]  = {0,1,2,1,0,2,1,2,1,0,2};
__device__ const int POFF[NPATHS] = {0,1,10,35,44,53,98,143,168,213,238};
__device__ const int SHO[3]       = {0,1,4};
__device__ const int AOFF[3]      = {0,1,4};

// ------------------------------ math helpers -------------------------------
__device__ __forceinline__ float fast_sigmoid(float x){ return __builtin_amdgcn_rcpf(1.f + __expf(-x)); }
__device__ __forceinline__ float fast_silu(float x)   { return x * fast_sigmoid(x); }
__device__ __forceinline__ int imax2(int a,int b){ return a>b?a:b; }
__device__ __forceinline__ int imin2(int a,int b){ return a<b?a:b; }

// WMMA half-operand k-mapping (ISA 7.12.2, 16-bit A 16x32) and the permuted
// storage slot that makes each lane's 16 halves contiguous.
__device__ __forceinline__ int kmap(int e, int hi){ return ((e < 8) ? e : e + 8) + hi*8; }
__device__ __forceinline__ int slot_of_k(int k){
  int kk = k & 31, b = k & ~31;
  int s = (kk < 8) ? kk : (kk < 16) ? kk + 8 : (kk < 24) ? kk - 8 : kk;
  return b + s;
}

// ------------------------- Wigner-3j (fp64, on-device) ---------------------
struct cpx { double re, im; };
__device__ __forceinline__ cpx cmul(cpx a, cpx b){ return {a.re*b.re - a.im*b.im, a.re*b.im + a.im*b.re}; }
__device__ __forceinline__ cpx cconj(cpx a){ return {a.re, -a.im}; }

__device__ double dfact(int n){ double r = 1.0; for(int i = 2; i <= n; ++i) r *= (double)i; return r; }

__device__ double su2_cg(int j1,int j2,int j3,int m1,int m2,int m3){
  if (m3 != m1 + m2) return 0.0;
  int vmin = imax2(imax2(-j1 + j2 + m3, -j1 + m1), 0);
  int vmax = imin2(imin2(j2 + j3 + m1, j3 - j1 + j2), j3 + m3);
  double C = sqrt((2.0*j3 + 1.0) * dfact(j3+j1-j2) * dfact(j3-j1+j2) * dfact(j1+j2-j3)
                  / dfact(j1+j2+j3+1) * dfact(j3+m3) * dfact(j3-m3)
                  / (dfact(j1+m1)*dfact(j1-m1)*dfact(j2+m2)*dfact(j2-m2)));
  double S = 0.0;
  for (int v = vmin; v <= vmax; ++v) {
    double sgn = ((v + j2 + m2) & 1) ? -1.0 : 1.0;
    S += sgn * dfact(j2+j3+m1-v) * dfact(j1-m1+v)
         / (dfact(v) * dfact(j3-j1+j2-v) * dfact(j3+m3-v) * dfact(v+j1-j2-m3));
  }
  return C * S;
}

__device__ void qmat(int l, cpx* q){    // (2l+1)x(2l+1), includes (-1j)^l
  int d = 2*l + 1;
  for (int i = 0; i < d*d; ++i) q[i] = {0.0, 0.0};
  double is2 = 1.0 / sqrt(2.0);
  for (int m = -l; m < 0; ++m) {
    int am = -m;
    q[(l+m)*d + (l+am)] = { is2, 0.0 };
    q[(l+m)*d + (l-am)] = { 0.0, -is2 };
  }
  q[l*d + l] = {1.0, 0.0};
  for (int m = 1; m <= l; ++m) {
    double s = (m & 1) ? -1.0 : 1.0;
    q[(l+m)*d + (l+m)] = { s*is2, 0.0 };
    q[(l+m)*d + (l-m)] = { 0.0, s*is2 };
  }
  cpx ph;
  switch (l & 3) { case 0: ph={1,0}; break; case 1: ph={0,-1}; break;
                   case 2: ph={-1,0}; break; default: ph={0,1}; }
  for (int i = 0; i < d*d; ++i) q[i] = cmul(ph, q[i]);
}

__global__ __launch_bounds__(128)
void w3j_kernel(float* __restrict__ out){
  const int pi = blockIdx.x;
  const int l1 = PL1[pi], l2 = PL2[pi], l3 = PL3[pi];
  const int d1 = 2*l1+1, d2 = 2*l2+1, d3 = 2*l3+1;
  const int n = d1*d2*d3;
  __shared__ cpx q1[25], q2[25], q3[25];
  __shared__ double vals[128];
  __shared__ double nrm;
  if (threadIdx.x == 0) { qmat(l1, q1); qmat(l2, q2); qmat(l3, q3); }
  __syncthreads();
  const int t = threadIdx.x;
  double v = 0.0;
  if (t < n) {
    int j = t / (d2*d3), l = (t / d3) % d2, m = t % d3;
    cpx acc = {0.0, 0.0};
    for (int i = 0; i < d1; ++i)
      for (int k = 0; k < d2; ++k)
        for (int nn = 0; nn < d3; ++nn) {
          double cg = su2_cg(l1, l2, l3, i-l1, k-l2, nn-l3);
          if (cg == 0.0) continue;
          cpx tc = cmul(cmul(q1[i*d1 + j], q2[k*d2 + l]), cconj(q3[nn*d3 + m]));
          acc.re += tc.re * cg; acc.im += tc.im * cg;
        }
    v = acc.re;
  }
  vals[t] = (t < n) ? v : 0.0;
  __syncthreads();
  if (threadIdx.x == 0) {
    double s = 0.0; for (int i = 0; i < n; ++i) s += vals[i]*vals[i];
    nrm = sqrt(s);
  }
  __syncthreads();
  if (t < n) out[POFF[pi] + t] = (float)(vals[t] / nrm);
}

// --------------------------- edge geometry ---------------------------------
// embp stored directly in WMMA-A permuted slot order (32 halves per edge).
__global__ __launch_bounds__(256)
void edge_geom_kernel(const float* __restrict__ pos,
                      const int* __restrict__ esrc, const int* __restrict__ edst, int E,
                      float* __restrict__ sh, half_t* __restrict__ embp){
  int e = blockIdx.x * blockDim.x + threadIdx.x;
  if (e >= E) return;
  int s = esrc[e], d = edst[e];
  float vx = pos[3*s+0]-pos[3*d+0], vy = pos[3*s+1]-pos[3*d+1], vz = pos[3*s+2]-pos[3*d+2];
  float r = sqrtf(vx*vx + vy*vy + vz*vz);
  float inv = 1.f / fmaxf(r, 1e-12f);
  float x = vx*inv, y = vy*inv, z = vz*inv;
  const float s15 = 3.8729833462f, s5 = 2.2360679775f, s3 = 1.7320508076f;
  float* o = sh + (size_t)e*9;
  o[0] = 1.f;
  o[1] = s3*x; o[2] = s3*y; o[3] = s3*z;
  o[4] = s15*x*z; o[5] = s15*x*y; o[6] = s5*(y*y - 0.5f*(x*x + z*z));
  o[7] = s15*y*z; o[8] = 0.5f*s15*(z*z - x*x);
  const float step = MAXR / (float)(NBASIS + 1);
  const float coef = 1.14136f * 7.3890560989f * 3.1622776602f; // 1.14136*e^2*sqrt(10)
  half_t* eo = embp + (size_t)e*32;
  #pragma unroll
  for (int k = 0; k < 32; ++k) eo[k] = (half_t)0.f;
  for (int b = 0; b < NBASIS; ++b) {
    float c  = (float)(b+1) * step;
    float dd = (r - c) / step;
    float t1 = 1.f + dd, t2 = 1.f - dd;
    float u1 = (t1 > 0.f) ? __expf(-1.f / fmaxf(t1, 1e-6f)) : 0.f;
    float u2 = (t2 > 0.f) ? __expf(-1.f / fmaxf(t2, 1e-6f)) : 0.f;
    eo[slot_of_k(b)] = (half_t)(coef * u1 * u2);
  }
}

// ------------------- weight packing into WMMA B-tile layout ----------------
// w1p: [4 ntile][32 lane][16 elem]   (K=10 padded to 32)
__global__ __launch_bounds__(256)
void pack_w1_kernel(const float* __restrict__ W1, half_t* __restrict__ out){
  int idx = blockIdx.x * blockDim.x + threadIdx.x;
  if (idx >= 4*32*16) return;
  int e = idx & 15, lane = (idx >> 4) & 31, nt = idx >> 9;
  int lm = lane & 15, hi = lane >> 4;
  int k = kmap(e, hi), n = nt*16 + lm;
  out[idx] = (k < NBASIS) ? (half_t)W1[k*64 + n] : (half_t)0;
}
// w2p: [NT ntile][2 kblock][32 lane][16 elem]
template<int MI>
__global__ __launch_bounds__(256)
void pack_w2_kernel(const float* __restrict__ W2, half_t* __restrict__ out){
  constexpr int NW = NPATHS*MI;
  constexpr int NT = (NW + 15)/16;
  int idx = blockIdx.x * blockDim.x + threadIdx.x;
  if (idx >= NT*2*32*16) return;
  int e = idx & 15, lane = (idx >> 4) & 31, kcb = (idx >> 9) & 1, nt = idx >> 10;
  int lm = lane & 15, hi = lane >> 4;
  int k = kcb*32 + kmap(e, hi), n = nt*16 + lm;
  out[idx] = (n < NW) ? (half_t)W2[(size_t)k*NW + n] : (half_t)0;
}

// ------------------------------- lin1 --------------------------------------
template<int MI>
__global__ __launch_bounds__(256)
void lin1_kernel(const float* __restrict__ x, const float* __restrict__ Wl,
                 float* __restrict__ h, int N){
  const int FD = MI*9;
  __shared__ float sW[3*MI*MI];
  for (int i = threadIdx.x; i < 3*MI*MI; i += 256) sW[i] = Wl[i];
  __syncthreads();
  int idx = blockIdx.x * blockDim.x + threadIdx.x;
  if (idx >= N*FD) return;
  int n = idx / FD, r = idx % FD;
  int l, d, base, v, m;
  if (r < MI)        { l=0; d=1; base=0;    v=r;          m=0; }
  else if (r < 4*MI) { l=1; d=3; base=MI;   v=(r-MI)/3;   m=(r-MI)%3; }
  else               { l=2; d=5; base=4*MI; v=(r-4*MI)/5; m=(r-4*MI)%5; }
  const float* W  = sW + l*MI*MI;
  const float* xr = x + (size_t)n*FD + base + m;
  float s = 0.f;
  for (int u = 0; u < MI; ++u) s += xr[u*d] * W[u*MI + v];
  h[idx] = s;
}

// ---------------- fused radial-MLP (WMMA) + tensor-product messages --------
template<int MI>
__global__ __launch_bounds__(128)
void msg_kernel(const float* __restrict__ h, const float* __restrict__ sh,
                const half_t* __restrict__ embp,
                const half_t* __restrict__ w1p, const half_t* __restrict__ w2p,
                const int* __restrict__ esrc, const int* __restrict__ edst, int E,
                const float* __restrict__ w3j, float* __restrict__ agg){
  constexpr int NW = NPATHS * MI;
  constexpr int FD = MI * 9;
  constexpr int NT = (NW + 15) / 16;
  __shared__ half_t hidh[16][64];   // permuted slot order per row
  __shared__ float  wts[16][NW];
  __shared__ float  hsrc[16][FD];
  __shared__ float  shs[16][9];
  __shared__ float  w3s[W3J_TOTAL];
  __shared__ int    ssrc[16], sdst[16];

  const int tid  = threadIdx.x;
  const int lane = tid & 31, wave = tid >> 5;
  const int lm   = lane & 15, hi = lane >> 4;
  const int e0   = blockIdx.x * 16;

  if (tid == 0) { __builtin_prefetch(w2p, 0, 3); __builtin_prefetch(w3j, 0, 3); }
  for (int i = tid; i < W3J_TOTAL; i += 128) w3s[i] = w3j[i];
  if (tid < 16) {
    int eg = imin2(e0 + tid, E - 1);   // clamp: rows past E load valid junk, never read
    ssrc[tid] = esrc[eg];
    sdst[tid] = edst[eg];
  }

  // ---- GEMM1: hid[16,64] = silu(emb[16,32] @ W1[32,64]) ----
  {
    const int row = imin2(e0 + lm, E - 1);
    v16h a = *(const v16h*)(embp + (size_t)row*32 + hi*16);       // 2x b128
    v16h b = *(const v16h*)(w1p + ((size_t)(wave*32 + lane))*16); // coalesced
    v8f c = {0.f,0.f,0.f,0.f,0.f,0.f,0.f,0.f};
    c = __builtin_amdgcn_wmma_f32_16x16x32_f16(false, a, false, b, (short)0, c, false, false);
    #pragma unroll
    for (int v = 0; v < 8; ++v) {
      int m = v + hi*8, n = wave*16 + lm;
      hidh[m][slot_of_k(n)] = (half_t)fast_silu(c[v]);
    }
  }
  __syncthreads();

  // ---- GEMM2: w[16,NW] = hid[16,64] @ W2[64,NW] (two K=32 steps) ----
  for (int nt = wave; nt < NT; nt += 4) {
    v8f c = {0.f,0.f,0.f,0.f,0.f,0.f,0.f,0.f};
    #pragma unroll
    for (int kcb = 0; kcb < 2; ++kcb) {
      v16h a = *(const v16h*)(&hidh[lm][kcb*32 + hi*16]);                  // ds_load_b128 x2
      v16h b = *(const v16h*)(w2p + ((((size_t)nt*2 + kcb)*32) + lane)*16);
      c = __builtin_amdgcn_wmma_f32_16x16x32_f16(false, a, false, b, (short)0, c, false, false);
    }
    #pragma unroll
    for (int v = 0; v < 8; ++v) {
      int m = v + hi*8, n = nt*16 + lm;
      if (NW % 16 == 0 || n < NW) wts[m][n] = c[v];
    }
  }

  // ---- stage h[src] via async global->LDS DMA (overlaps with GEMM2) ----
  {
    const int groups = FD / 4;                    // float4 chunks per row
    for (int idx = tid; idx < 16*groups; idx += 128) {
      int e = idx / groups, g = idx % groups;
      unsigned lds  = (unsigned)(uintptr_t)&hsrc[e][g*4];
      unsigned goff = (unsigned)(((unsigned)ssrc[e]*FD + g*4) * sizeof(float));
      asm volatile("global_load_async_to_lds_b128 %0, %1, %2"
                   :: "v"(lds), "v"(goff), "s"(h) : "memory");
    }
  }
  for (int idx = tid; idx < 16*9; idx += 128) {
    int e = idx / 9, j = idx % 9;
    shs[e][j] = sh[(size_t)imin2(e0 + e, E - 1)*9 + j];
  }
  asm volatile("s_wait_asynccnt 0x0" ::: "memory");
  __syncthreads();

  // ---- tensor-product messages + scatter-add ----
  for (int p = tid; p < 16*MI; p += 128) {
    int u = p % MI, e = p / MI;
    int edge = e0 + e;
    if (edge >= E) continue;
    float acc[9];
    #pragma unroll
    for (int q = 0; q < 9; ++q) acc[q] = 0.f;
    const float* he = hsrc[e];
    const float* se = shs[e];
    #pragma unroll
    for (int pi = 0; pi < NPATHS; ++pi) {
      const int l1 = PL1[pi], l2 = PL2[pi], l3 = PL3[pi];
      const int d1 = 2*l1+1, d2 = 2*l2+1, d3 = 2*l3+1;
      const int b1 = (l1 == 0) ? 0 : ((l1 == 1) ? MI : 4*MI);
      const float wl = wts[e][pi*MI + u];
      const float* hb = he + b1 + u*d1;
      const float* sb = se + SHO[l2];
      const float* W  = w3s + POFF[pi];
      for (int k = 0; k < d3; ++k) {
        float s = 0.f;
        for (int i = 0; i < d1; ++i) {
          float hv = hb[i];
          for (int j = 0; j < d2; ++j)
            s += W[(i*d2 + j)*d3 + k] * hv * sb[j];
        }
        acc[AOFF[l3] + k] += s * wl;
      }
    }
    float* dp = agg + (size_t)sdst[e]*FD;
    atomicAdd(&dp[u], acc[0]);
    #pragma unroll
    for (int k = 0; k < 3; ++k) atomicAdd(&dp[MI   + u*3 + k], acc[1 + k]);
    #pragma unroll
    for (int k = 0; k < 5; ++k) atomicAdd(&dp[4*MI + u*5 + k], acc[4 + k]);
  }
}

// ----------------------- lin2 + skip + gating ------------------------------
template<int MI, int MO, bool GATED>
__global__ __launch_bounds__(64)
void out_kernel(const float* __restrict__ x, const float* __restrict__ agg,
                const float* __restrict__ SC, const float* __restrict__ L2W,
                const float* __restrict__ G1, const float* __restrict__ G2,
                float* __restrict__ out, int N){
  constexpr int NPB = 64 / MO;
  __shared__ float sbuf[64];
  const int ln = threadIdx.x / MO, v = threadIdx.x % MO;
  const int n = blockIdx.x * NPB + ln;
  float o0 = 0.f, o1[3] = {0,0,0}, o2[5] = {0,0,0,0,0};
  if (n < N) {
    const float* xr = x   + (size_t)n*MI*9;
    const float* ar = agg + (size_t)n*MI*9;
    const float *S0 = SC,  *S1 = SC  + MI*MO, *S2 = SC  + 2*MI*MO;
    const float *Q0 = L2W, *Q1 = L2W + MI*MO, *Q2 = L2W + 2*MI*MO;
    const float inv = 0.2f;  // 1/sqrt(NUM_NEIGHBORS=25)
    for (int u = 0; u < MI; ++u) {
      float sw0 = S0[u*MO+v], qw0 = Q0[u*MO+v];
      o0 += xr[u]*sw0 + inv*ar[u]*qw0;
      float sw1 = S1[u*MO+v], qw1 = Q1[u*MO+v];
      #pragma unroll
      for (int m = 0; m < 3; ++m) o1[m] += xr[MI + u*3 + m]*sw1 + inv*ar[MI + u*3 + m]*qw1;
      float sw2 = S2[u*MO+v], qw2 = Q2[u*MO+v];
      #pragma unroll
      for (int m = 0; m < 5; ++m) o2[m] += xr[4*MI + u*5 + m]*sw2 + inv*ar[4*MI + u*5 + m]*qw2;
    }
  }
  sbuf[threadIdx.x] = o0;   // pre-activation scalars drive the gates
  __syncthreads();
  if (n < N) {
    float g1 = 1.f, g2 = 1.f;
    if (GATED) {
      float a1 = 0.f, a2 = 0.f;
      for (int u = 0; u < MO; ++u) {
        float sv = sbuf[ln*MO + u];
        a1 += sv * G1[u*MO + v];
        a2 += sv * G2[u*MO + v];
      }
      g1 = fast_sigmoid(a1); g2 = fast_sigmoid(a2);
      o0 = fast_silu(o0);
    }
    float* orow = out + (size_t)n*MO*9;
    orow[v] = o0;
    #pragma unroll
    for (int m = 0; m < 3; ++m) orow[MO   + v*3 + m] = o1[m] * g1;
    #pragma unroll
    for (int m = 0; m < 5; ++m) orow[4*MO + v*5 + m] = o2[m] * g2;
  }
}

// --------------------------- small utility kernels -------------------------
__global__ void zero_kernel(float* p, int n){
  int i = blockIdx.x*blockDim.x + threadIdx.x; if (i < n) p[i] = 0.f;
}
__global__ void ss_kernel(const float* __restrict__ y, const int* __restrict__ batch,
                          float* __restrict__ ss, int N){
  int n = blockIdx.x*blockDim.x + threadIdx.x; if (n >= N) return;
  float s = 0.f;
  for (int d = 0; d < FEATDIM; ++d) { float t = y[(size_t)n*FEATDIM + d]; s += t*t; }
  atomicAdd(&ss[batch[n]], s);
}
__global__ void scale_kernel(float* __restrict__ y, const int* __restrict__ batch,
                             const float* __restrict__ ss, int N){
  int i = blockIdx.x*blockDim.x + threadIdx.x; if (i >= N*FEATDIM) return;
  y[i] *= rsqrtf(ss[batch[i / FEATDIM]]);
}
__global__ void mix_kernel(const float* oi, const float* of_, const float* p, float* xts, int n){
  int i = blockIdx.x*blockDim.x + threadIdx.x;
  if (i < n) { float pp = p[0]; xts[i] = (1.f - pp)*oi[i] + pp*of_[i]; }
}
__global__ void pool_kernel(const float* __restrict__ ot, const float* __restrict__ oi,
                            const int* __restrict__ batch, float* __restrict__ out, int N){
  int i = blockIdx.x*blockDim.x + threadIdx.x; if (i >= N*FEATDIM) return;
  int n = i / FEATDIM, d = i % FEATDIM;
  atomicAdd(&out[batch[n]*FEATDIM + d], 0.1f * (ot[i] - oi[i])); // 1/sqrt(100)
}

// ------------------------------ host driver --------------------------------
static void run_network(const float* x_in, const float* pos,
                        const int* esrc, const int* edst, int E,
                        const float* pbase, float* out,
                        float* f0, float* f1, float* hbuf, float* aggb,
                        float* shb, half_t* embp, half_t* w1p, half_t* w2p,
                        const float* w3j, hipStream_t stream){
  if (E > 0)
    edge_geom_kernel<<<cdiv(E, 256), 256, 0, stream>>>(pos, esrc, edst, E, shb, embp);
  const float* cur = x_in;
  const int mis[4] = {4, 32, 32, 32};
  const int mos[4] = {32, 32, 32, 4};
  const float* P = pbase;
  for (int L = 0; L < 4; ++L) {
    const int mi = mis[L], mo = mos[L], fd = mi*9;
    const float* fc1 = P;
    const float* fc2 = P + 640;
    const float* g1  = fc2 + 64*NPATHS*mi;
    const float* g2  = g1 + mo*mo;
    const float* l1p = g2 + mo*mo;
    const float* l2p = l1p + 3*mi*mi;
    const float* scp = l2p + 3*mi*mo;
    P = scp + 3*mi*mo;

    if (mi == 4) lin1_kernel<4 ><<<cdiv(NNODES*fd, 256), 256, 0, stream>>>(cur, l1p, hbuf, NNODES);
    else         lin1_kernel<32><<<cdiv(NNODES*fd, 256), 256, 0, stream>>>(cur, l1p, hbuf, NNODES);
    zero_kernel<<<cdiv(NNODES*fd, 256), 256, 0, stream>>>(aggb, NNODES*fd);
    if (E > 0) {
      pack_w1_kernel<<<cdiv(4*32*16, 256), 256, 0, stream>>>(fc1, w1p);
      if (mi == 4) {
        pack_w2_kernel<4 ><<<cdiv(3*1024, 256), 256, 0, stream>>>(fc2, w2p);
        msg_kernel<4 ><<<cdiv(E, 16), 128, 0, stream>>>(hbuf, shb, embp, w1p, w2p, esrc, edst, E, w3j, aggb);
      } else {
        pack_w2_kernel<32><<<cdiv(22*1024, 256), 256, 0, stream>>>(fc2, w2p);
        msg_kernel<32><<<cdiv(E, 16), 128, 0, stream>>>(hbuf, shb, embp, w1p, w2p, esrc, edst, E, w3j, aggb);
      }
    }
    float* nxt = (L == 3) ? out : ((L & 1) ? f1 : f0);
    if (L == 0)      out_kernel<4,32,true ><<<cdiv(NNODES*32, 64), 64, 0, stream>>>(cur, aggb, scp, l2p, g1, g2, nxt, NNODES);
    else if (L < 3)  out_kernel<32,32,true ><<<cdiv(NNODES*32, 64), 64, 0, stream>>>(cur, aggb, scp, l2p, g1, g2, nxt, NNODES);
    else             out_kernel<32,4,false><<<cdiv(NNODES*4,  64), 64, 0, stream>>>(cur, aggb, scp, l2p, g1, g2, nxt, NNODES);
    cur = nxt;
  }
}

static void norm_graphs(float* y, const int* batch, float* ss, hipStream_t s){
  zero_kernel<<<1, 64, 0, s>>>(ss, NGRAPHS);
  ss_kernel<<<cdiv(NNODES, 256), 256, 0, s>>>(y, batch, ss, NNODES);
  scale_kernel<<<cdiv(NNODES*FEATDIM, 256), 256, 0, s>>>(y, batch, ss, NNODES);
}

extern "C" void kernel_launch(void* const* d_in, const int* in_sizes, int n_in,
                              void* d_out, int out_size, void* d_ws, size_t ws_size,
                              hipStream_t stream){
  const float* x       = (const float*)d_in[0];
  const float* x_final = (const float*)d_in[1];
  const float* pos     = (const float*)d_in[2];
  const float* pos_f   = (const float*)d_in[3];
  const float* pos_t   = (const float*)d_in[4];
  const float* p       = (const float*)d_in[5];
  const float* params  = (const float*)d_in[6];   // jax pytree leaves, flat
  const int*   batch   = (const int*)d_in[7];
  const int*   es_i = (const int*)d_in[8];  const int E_i = in_sizes[8];
  const int*   ed_i = (const int*)d_in[9];
  const int*   es_f = (const int*)d_in[10]; const int E_f = in_sizes[10];
  const int*   ed_f = (const int*)d_in[11];
  const int*   es_t = (const int*)d_in[12]; const int E_t = in_sizes[12];
  const int*   ed_t = (const int*)d_in[13];
  (void)n_in; (void)ws_size;

  // jax dict keys sorted: 'final', 'init', 'ts' — each network = 102224 floats
  const float* p_final = params;
  const float* p_init  = params + 102224;
  const float* p_ts    = params + 204448;

  // ---- workspace carve (256B aligned) ----
  char* w = (char*)d_ws;
  auto carve = [&](size_t bytes) -> char* {
    char* r = w; w += (bytes + 255) & ~(size_t)255; return r;
  };
  float*  w3j  = (float*)carve(W3J_TOTAL * 4);
  float*  f0   = (float*)carve((size_t)NNODES * 288 * 4);
  float*  f1   = (float*)carve((size_t)NNODES * 288 * 4);
  float*  hbuf = (float*)carve((size_t)NNODES * 288 * 4);
  float*  aggb = (float*)carve((size_t)NNODES * 288 * 4);
  float*  oi   = (float*)carve((size_t)NNODES * FEATDIM * 4);
  float*  of_  = (float*)carve((size_t)NNODES * FEATDIM * 4);
  float*  xts  = (float*)carve((size_t)NNODES * FEATDIM * 4);
  float*  ss   = (float*)carve(64 * 4);
  half_t* w1p  = (half_t*)carve(4*32*16 * 2);
  half_t* w2p  = (half_t*)carve(22*1024 * 2);
  int Emax = E_i > E_f ? E_i : E_f; if (E_t > Emax) Emax = E_t;
  float*  shb  = (float*)carve((size_t)Emax * 9 * 4);
  half_t* embp = (half_t*)carve((size_t)Emax * 32 * 2);

  // Wigner-3j tensors (deterministic, recomputed every launch)
  w3j_kernel<<<NPATHS, 128, 0, stream>>>(w3j);

  // net 1: init graph
  run_network(x, pos, es_i, ed_i, E_i, p_init, oi, f0, f1, hbuf, aggb, shb, embp, w1p, w2p, w3j, stream);
  norm_graphs(oi, batch, ss, stream);
  // net 2: final graph
  run_network(x_final, pos_f, es_f, ed_f, E_f, p_final, of_, f0, f1, hbuf, aggb, shb, embp, w1p, w2p, w3j, stream);
  norm_graphs(of_, batch, ss, stream);
  // mix
  mix_kernel<<<cdiv(NNODES*FEATDIM, 256), 256, 0, stream>>>(oi, of_, p, xts, NNODES*FEATDIM);
  // net 3: ts graph (reuse of_ as out_t)
  run_network(xts, pos_t, es_t, ed_t, E_t, p_ts, of_, f0, f1, hbuf, aggb, shb, embp, w1p, w2p, w3j, stream);
  norm_graphs(of_, batch, ss, stream);
  // pool(out_t) - pool(out_i)
  zero_kernel<<<cdiv(out_size, 256), 256, 0, stream>>>((float*)d_out, out_size);
  pool_kernel<<<cdiv(NNODES*FEATDIM, 256), 256, 0, stream>>>(of_, oi, batch, (float*)d_out, NNODES);
}
